// TreeEncoder_16458314678305
// MI455X (gfx1250) — matrix operand order repaired and checked by reading, not compiled, not run on hardware
//
#include <hip/hip_runtime.h>

// ---------------------------------------------------------------------------
// TreeEncoder on MI455X (gfx1250): bf16 WMMA GEMMs with fp32 accumulate.
// Conv: M=32 x N=128 tiles, async-to-LDS bf16 gather, direct fp32 writeback.
// ---------------------------------------------------------------------------

typedef __attribute__((ext_vector_type(16))) __bf16 v16bf;
typedef __attribute__((ext_vector_type(8)))  float  v8f;

union Frag16 { v16bf v; uint4 q[2]; };

#define MAXD 9
#define HID 128

__device__ __forceinline__ unsigned mort_ileave(unsigned x) {
  x &= 0xFFFFu;
  x = (x | (x << 8)) & 0x00FF00FFu;
  x = (x | (x << 4)) & 0x0F0F0F0Fu;
  x = (x | (x << 2)) & 0x33333333u;
  x = (x | (x << 1)) & 0x55555555u;
  return x;
}
__device__ __forceinline__ unsigned mort_dileave(unsigned x) {
  x &= 0x55555555u;
  x = (x | (x >> 1)) & 0x33333333u;
  x = (x | (x >> 2)) & 0x0F0F0F0Fu;
  x = (x | (x >> 4)) & 0x00FF00FFu;
  x = (x | (x >> 8)) & 0x0000FFFFu;
  return x;
}
__device__ __forceinline__ unsigned short f32_bf16(float f) {
  unsigned u = __float_as_uint(f);
  u += 0x7FFFu + ((u >> 16) & 1u);   // round-to-nearest-even
  return (unsigned short)(u >> 16);
}
__device__ __forceinline__ unsigned pack_bf16x2(float a, float b) {
  return (unsigned)f32_bf16(a) | ((unsigned)f32_bf16(b) << 16);
}

// ---------------------------------------------------------------------------
// Kernel 0: convert conv_W and emb_W (fp32) to bf16 in workspace (L2-resident)
// ---------------------------------------------------------------------------
__global__ __launch_bounds__(256) void prep_weights(
    const float* __restrict__ convW, const float* __restrict__ embW,
    unsigned short* __restrict__ convWb, unsigned short* __restrict__ embWb,
    int n_conv, int n_total) {
  int i = blockIdx.x * 256 + threadIdx.x;
  if (i >= n_total) return;
  if (i < n_conv) convWb[i] = f32_bf16(convW[i]);
  else            embWb[i - n_conv] = f32_bf16(embW[i - n_conv]);
}

// ---------------------------------------------------------------------------
// Kernel 0b: snapshot one level of h (fp32) into a bf16 copy for the gather.
// ---------------------------------------------------------------------------
__global__ __launch_bounds__(256) void cvt_level_bf16(
    const float* __restrict__ src, unsigned short* __restrict__ dst, long n4) {
  long q = (long)blockIdx.x * 256 + threadIdx.x;   // float4 units
  if (q >= n4) return;
  float4 v = ((const float4*)src)[q];
  uint2 pk;
  pk.x = pack_bf16x2(v.x, v.y);
  pk.y = pack_bf16x2(v.z, v.w);
  ((uint2*)dst)[q] = pk;
}

// ---------------------------------------------------------------------------
// Kernel 1: in_proj. One block (128 thr) per node: build 40-dim input in LDS,
// each thread computes one of 128 output channels.
// ---------------------------------------------------------------------------
__global__ __launch_bounds__(128) void in_proj_kernel(
    const float* __restrict__ feats, const float* __restrict__ W,
    const float* __restrict__ b, float* __restrict__ h) {
  __shared__ float xin[40];
  const int g = blockIdx.x;
  int d = 0; long off = 0;
  for (int dd = 0; dd <= MAXD; ++dd) {
    long sz = 1L << (2 * dd);
    if (g < off + sz) { d = dd; break; }
    off += sz;
  }
  const int t = threadIdx.x;
  if (t < 40) {
    float val;
    if (t == 0) {
      val = feats[g];
    } else {
      unsigned local = (unsigned)(g - off);
      unsigned ix = mort_dileave(local), iy = mort_dileave(local >> 1);
      float res = (float)(1 << d);
      float px = (ix + 0.5f) / res, py = (iy + 0.5f) / res;
      float pdn = (float)d / (float)MAXD;
      if (t == 1)      val = px;
      else if (t == 2) val = py;
      else if (t == 3) val = pdn;
      else {
        int i = t - 4;          // 0..35
        int k = i / 12, rem = i - 12 * k;
        float p = (k == 0) ? px : ((k == 1) ? py : pdn);
        float ang = p * 6.28318530717958647f * (float)(1 << (rem % 6));
        val = (rem < 6) ? __sinf(ang) : __cosf(ang);
      }
    }
    xin[t] = val;
  }
  __syncthreads();
  const float* wr = W + t * 40;
  float acc = b[t];
  #pragma unroll
  for (int k = 0; k < 40; ++k) acc = fmaf(wr[k], xin[k], acc);
  h[(long)g * HID + t] = acc;
}

// ---------------------------------------------------------------------------
// Kernel 2: pool children (depth d) into parents (depth d-1): mean of 4.
// float4-vectorized: children rows are 32-float4 strides apart.
// ---------------------------------------------------------------------------
__global__ __launch_bounds__(256) void pool_kernel(
    float* __restrict__ h, long off_par, long off_ch, int n_par) {
  long idx = (long)blockIdx.x * 256 + threadIdx.x;   // float4 units
  long n4 = (long)n_par * 32;
  if (idx >= n4) return;
  long p = idx >> 5; int q = (int)(idx & 31);
  const float4* cb = (const float4*)(h + (off_ch + p * 4) * HID) + q;
  float4 a = cb[0], b = cb[32], c = cb[64], d = cb[96];
  float4* pp = (float4*)(h + (off_par + p) * HID) + q;
  float4 r = *pp;
  r.x += 0.25f * (a.x + b.x + c.x + d.x);
  r.y += 0.25f * (a.y + b.y + c.y + d.y);
  r.z += 0.25f * (a.z + b.z + c.z + d.z);
  r.w += 0.25f * (a.w + b.w + c.w + d.w);
  *pp = r;
}

// ---------------------------------------------------------------------------
// Kernel 3: quadconv at depth dd. Block = 256 thr (8 waves), tile = 32 nodes.
// Gather: neighbor keys precomputed, then GLOBAL_LOAD_ASYNC_TO_LDS_B64 pulls
// bf16 activation rows cache->LDS (no VGPR round trip, ASYNCcnt tracked).
// Wave w computes output channels [16w,16w+16) for all 32 rows with two
// independent WMMA accumulator chains. Output staged in LDS (aliasing Xs),
// written back as coalesced b128 stores DIRECTLY into h (gather reads the
// bf16 snapshot, so no tmp buffer / copy-back is needed).
// ---------------------------------------------------------------------------
__global__ __launch_bounds__(256) void quadconv_kernel(
    const unsigned short* __restrict__ hb,   // bf16 snapshot of level [N][128]
    const unsigned short* __restrict__ Wb,   // bf16 [128][1152]
    const float* __restrict__ bias,          // [128]
    float* __restrict__ out,                 // h + OFF[dd]*HID, [N][128] f32
    int N, int dd) {
  __shared__ __align__(16) unsigned short Xs[32][9 * HID];   // 72 KB (also Os)
  __shared__ int nkey[32 * 9];
  const int res = 1 << dd;
  const int tile0 = blockIdx.x * 32;
  const int tid = threadIdx.x;

  // neighbor keys for 32 rows x 9 taps (-1 = zero row)
  for (int i = tid; i < 32 * 9; i += 256) {
    int row = i / 9, j = i - row * 9;
    int node = tile0 + row;
    int nk = -1;
    if (node < N) {
      unsigned key = (unsigned)node;
      int ix = (int)mort_dileave(key), iy = (int)mort_dileave(key >> 1);
      int nx = ix + (j % 3) - 1, ny = iy + (j / 3) - 1;
      if (nx >= 0 && nx < res && ny >= 0 && ny < res)
        nk = (int)(mort_ileave((unsigned)nx) | (mort_ileave((unsigned)ny) << 1));
    }
    nkey[i] = nk;
  }
  __syncthreads();

  // gather: each 32-lane group pulls one (row,tap) = 256 B of bf16 into LDS
  // via async copy (8 B / lane). nk is uniform within the group.
  {
    const int lane = tid & 31;
    for (int p = tid >> 5; p < 32 * 9; p += 8) {
      int row = p / 9, j = p - row * 9;
      int nk = nkey[p];
      unsigned short* ldsp = &Xs[row][j * HID + lane * 4];
      if (nk >= 0) {
        unsigned lds_off = (unsigned)(size_t)(void*)ldsp;
        unsigned long long ga =
            (unsigned long long)(size_t)(hb + (long)nk * HID + lane * 4);
        asm volatile("global_load_async_to_lds_b64 %0, %1, off"
                     :: "v"(lds_off), "v"(ga) : "memory");
      } else {
        *(uint2*)ldsp = make_uint2(0u, 0u);
      }
    }
  }
  asm volatile("s_wait_asynccnt 0x0" ::: "memory");
  __syncthreads();

  const int wave = tid >> 5;
  const int lane = tid & 31;
  const int mrow = lane & 15;
  const int ncol = wave * 16 + mrow;
  const int a_off = (lane < 16) ? 0 : 8;    // A K-sub-offset
  const int b_off = (lane < 16) ? 0 : 8;    // B K-sub-offset
  const unsigned short* wrow = Wb + (long)ncol * (9 * HID);
  const unsigned short* xr0 = &Xs[mrow][0];
  const unsigned short* xr1 = &Xs[16 + mrow][0];

  v8f acc0 = {0.f, 0.f, 0.f, 0.f, 0.f, 0.f, 0.f, 0.f};
  v8f acc1 = {0.f, 0.f, 0.f, 0.f, 0.f, 0.f, 0.f, 0.f};
  #pragma unroll 2
  for (int kk = 0; kk < 36; ++kk) {
    int kb = kk * 32;
    Frag16 b, a0, a1;
    b.q[0] = *(const uint4*)(wrow + kb + b_off);
    b.q[1] = *(const uint4*)(wrow + kb + b_off + 16);
    a0.q[0] = *(const uint4*)(xr0 + kb + a_off);
    a0.q[1] = *(const uint4*)(xr0 + kb + a_off + 16);
    a1.q[0] = *(const uint4*)(xr1 + kb + a_off);
    a1.q[1] = *(const uint4*)(xr1 + kb + a_off + 16);
    acc0 = __builtin_amdgcn_wmma_f32_16x16x32_bf16(
        false, a0.v, false, b.v, (short)0, acc0, false, false);
    acc1 = __builtin_amdgcn_wmma_f32_16x16x32_bf16(
        false, a1.v, false, b.v, (short)0, acc1, false, false);
  }
  __syncthreads();   // all Xs reads done; safe to alias as output stage

  float* Os = (float*)&Xs[0][0];           // 32 x 128 f32 = 16 KB
  const float bv = bias[ncol];
  const int row0 = (lane < 16) ? 0 : 8;
  #pragma unroll
  for (int r = 0; r < 8; ++r) {
    Os[(row0 + r) * HID + ncol]      = fmaxf(acc0[r] + bv, 0.0f);
    Os[(16 + row0 + r) * HID + ncol] = fmaxf(acc1[r] + bv, 0.0f);
  }
  __syncthreads();

  // coalesced b128 stores: 32 rows x 128 ch = 1024 float4
  const float4* Osv = (const float4*)Os;
  float4* outv = (float4*)(out + (long)tile0 * HID);
  const bool full = (tile0 + 32 <= N);
  #pragma unroll
  for (int it = 0; it < 4; ++it) {
    int q = it * 256 + tid;                // float4 index, row = q/32
    if (full || tile0 + (q >> 5) < N) outv[q] = Osv[q];
  }
}

// ---------------------------------------------------------------------------
// Kernel 4: emb GEMM (128->128, 2x2 WMMA chains/wave) + LayerNorm + gain.
// One launch covers all depths; block -> (depth, 16-node tile) via prefix scan.
// ---------------------------------------------------------------------------
__global__ __launch_bounds__(256) void emb_ln_kernel(
    const float* __restrict__ h, const unsigned short* __restrict__ embWb,
    const float* __restrict__ emb_b, const float* __restrict__ ln_g,
    const float* __restrict__ ln_b, const float* __restrict__ gain,
    float* __restrict__ out) {
  __shared__ __align__(16) unsigned short Xs[16][HID];
  __shared__ float Zs[16][HID];
  __shared__ float s_mu[16], s_rs[16];

  // map blockIdx -> (depth d, tile)
  int b = blockIdx.x, d = 0, tile = 0, sz = 1;
  long off = 0;
  {
    int acc = 0;
    for (int dd = 0; dd <= MAXD; ++dd) {
      int s = 1 << (2 * dd);
      int td = (s + 15) >> 4;
      if (b < acc + td) { d = dd; tile = b - acc; sz = s; break; }
      acc += td; off += s;
    }
  }
  const int tid = threadIdx.x;

  // stage activations: 16 rows x 32 float4 -> bf16
  for (int q = tid; q < 16 * 32; q += 256) {
    int row = q >> 5, f4 = q & 31;
    int ln = tile * 16 + row;
    float4 v = make_float4(0.f, 0.f, 0.f, 0.f);
    if (ln < sz) v = ((const float4*)(h + (off + ln) * (long)HID))[f4];
    uint2 pk;
    pk.x = pack_bf16x2(v.x, v.y);
    pk.y = pack_bf16x2(v.z, v.w);
    *(uint2*)&Xs[row][f4 * 4] = pk;
  }
  __syncthreads();

  const int wave = tid >> 5;
  const int lane = tid & 31;
  const int mrow = lane & 15;
  const int ncol = wave * 16 + mrow;
  const int a_off = (lane < 16) ? 0 : 8;
  const int b_off = (lane < 16) ? 0 : 8;
  const unsigned short* wrow = embWb + ((long)d * HID + ncol) * HID;

  v8f acc0 = {0.f, 0.f, 0.f, 0.f, 0.f, 0.f, 0.f, 0.f};
  v8f acc1 = {0.f, 0.f, 0.f, 0.f, 0.f, 0.f, 0.f, 0.f};
  #pragma unroll
  for (int kk = 0; kk < 2; ++kk) {
    int kb0 = kk * 32, kb1 = (kk + 2) * 32;
    Frag16 a0, b0, a1, b1;
    a0.q[0] = *(const uint4*)&Xs[mrow][kb0 + a_off];
    a0.q[1] = *(const uint4*)&Xs[mrow][kb0 + a_off + 16];
    b0.q[0] = *(const uint4*)(wrow + kb0 + b_off);
    b0.q[1] = *(const uint4*)(wrow + kb0 + b_off + 16);
    a1.q[0] = *(const uint4*)&Xs[mrow][kb1 + a_off];
    a1.q[1] = *(const uint4*)&Xs[mrow][kb1 + a_off + 16];
    b1.q[0] = *(const uint4*)(wrow + kb1 + b_off);
    b1.q[1] = *(const uint4*)(wrow + kb1 + b_off + 16);
    acc0 = __builtin_amdgcn_wmma_f32_16x16x32_bf16(
        false, a0.v, false, b0.v, (short)0, acc0, false, false);
    acc1 = __builtin_amdgcn_wmma_f32_16x16x32_bf16(
        false, a1.v, false, b1.v, (short)0, acc1, false, false);
  }
  v8f accs = acc0 + acc1;

  const float bv = emb_b[d * HID + ncol];
  const int row0 = (lane < 16) ? 0 : 8;
  #pragma unroll
  for (int r = 0; r < 8; ++r) Zs[row0 + r][ncol] = accs[r] + bv;
  __syncthreads();

  if (tid < 16) {
    int row = tid;
    float s = 0.f, ss = 0.f;
    #pragma unroll 8
    for (int c = 0; c < HID; ++c) { float z = Zs[row][c]; s += z; ss += z * z; }
    float mu = s * (1.0f / HID);
    float var = ss * (1.0f / HID) - mu * mu;
    s_mu[row] = mu;
    s_rs[row] = rsqrtf(var + 1e-5f);
  }
  __syncthreads();

  const float dg = gain[d];
  for (int idx = tid; idx < 16 * HID; idx += 256) {
    int row = idx >> 7, c = idx & 127;
    int ln = tile * 16 + row;
    if (ln < sz) {
      float z = (Zs[row][c] - s_mu[row]) * s_rs[row];
      out[(off + ln) * (long)HID + c] = dg * (z * ln_g[d * HID + c] + ln_b[d * HID + c]);
    }
  }
}

// ---------------------------------------------------------------------------
// Host orchestration
// ---------------------------------------------------------------------------
extern "C" void kernel_launch(void* const* d_in, const int* in_sizes, int n_in,
                              void* d_out, int out_size, void* d_ws, size_t ws_size,
                              hipStream_t stream) {
  static const int  SZ[10]  = {1, 4, 16, 64, 256, 1024, 4096, 16384, 65536, 262144};
  static const long OFF[10] = {0, 1, 5, 21, 85, 341, 1365, 5461, 21845, 87381};
  const long TOTAL = 349525;

  const float* feats     = (const float*)d_in[0];
  const float* in_proj_W = (const float*)d_in[1];
  const float* in_proj_b = (const float*)d_in[2];
  const float* conv_W    = (const float*)d_in[3];
  const float* conv_b    = (const float*)d_in[4];
  const float* emb_W     = (const float*)d_in[5];
  const float* emb_b     = (const float*)d_in[6];
  const float* ln_g      = (const float*)d_in[7];
  const float* ln_b      = (const float*)d_in[8];
  const float* gain      = (const float*)d_in[9];
  float* out = (float*)d_out;

  // workspace layout
  const long H_ELEMS     = TOTAL * HID;          // 44,739,200 f32
  const long HB_ELEMS    = 65536L * HID;         //  8,388,608 slots (bf16 used)
  const long CONVW_ELEMS = 10L * 128 * 1152;     //  1,474,560 bf16
  const long EMBW_ELEMS  = 10L * 128 * 128;      //    163,840 bf16
  float* h = (float*)d_ws;
  unsigned short* hb     = (unsigned short*)(h + H_ELEMS);   // level snapshot
  unsigned short* convWb = hb + 2 * HB_ELEMS;    // keep old tmp footprint
  unsigned short* embWb  = convWb + CONVW_ELEMS;

  // 0: weight conversion to bf16
  {
    int n_total = (int)(CONVW_ELEMS + EMBW_ELEMS);
    prep_weights<<<(n_total + 255) / 256, 256, 0, stream>>>(
        conv_W, emb_W, convWb, embWb, (int)CONVW_ELEMS, n_total);
  }

  // 1: in_proj for all nodes
  in_proj_kernel<<<(unsigned)TOTAL, 128, 0, stream>>>(feats, in_proj_W, in_proj_b, h);

  // 2: top-down pool + quadconv
  for (int d = MAXD; d >= 1; --d) {
    int n_par = SZ[d - 1];
    long n4 = (long)n_par * 32;                  // float4 units
    pool_kernel<<<(unsigned)((n4 + 255) / 256), 256, 0, stream>>>(
        h, OFF[d - 1], OFF[d], n_par);
    int dd = d - 1;
    if (dd >= 1) {
      int N = SZ[dd];
      long lvl4 = (long)N * 32;
      cvt_level_bf16<<<(unsigned)((lvl4 + 255) / 256), 256, 0, stream>>>(
          h + OFF[dd] * HID, hb, lvl4);
      quadconv_kernel<<<(N + 31) / 32, 256, 0, stream>>>(
          hb, convWb + (long)dd * 128 * 1152, conv_b + dd * 128,
          h + OFF[dd] * HID, N, dd);
    }
  }

  // 3: emb + layernorm + gain -> out
  {
    int tiles = 0;
    for (int d = 0; d <= MAXD; ++d) tiles += (SZ[d] + 15) >> 4;   // 21847
    emb_ln_kernel<<<tiles, 256, 0, stream>>>(h, embWb, emb_b, ln_g, ln_b, gain, out);
  }
}
